// batched_MGGP_RBF_37048387895520
// MI455X (gfx1250) — matrix operand order repaired and checked by reading, compiled, not run on hardware
//
#include <hip/hip_runtime.h>
#include <math.h>

typedef __attribute__((ext_vector_type(16))) __bf16 v16bf;
typedef __attribute__((ext_vector_type(8)))  __bf16 v8bf;
typedef __attribute__((ext_vector_type(4)))  __bf16 v4bf;
typedef __attribute__((ext_vector_type(8)))  float  v8f;

#define BM 128
#define BN 128
#define BK 32
#define LDT 40  // padded tile row stride in bf16 elems (80 B: 16B-aligned, reduces bank conflicts)

// --- precompute per-row squared norms: one wave per row ---
__global__ __launch_bounds__(256) void row_sumsq_kernel(const float* __restrict__ A,
                                                        float* __restrict__ out,
                                                        int rows, int D) {
  const int row  = blockIdx.x * 8 + (threadIdx.x >> 5);
  const int lane = threadIdx.x & 31;
  if (row >= rows) return;
  const float* p = A + (size_t)row * D;
  float s = 0.f;
  for (int k = lane * 4; k + 3 < D; k += 128) {
    float4 v = *(const float4*)(p + k);
    s = fmaf(v.x, v.x, s); s = fmaf(v.y, v.y, s);
    s = fmaf(v.z, v.z, s); s = fmaf(v.w, v.w, s);
  }
#pragma unroll
  for (int off = 16; off > 0; off >>= 1) s += __shfl_xor(s, off, 32);
  if (lane == 0) out[row] = s;
}

__device__ __forceinline__ v16bf cat8(v8bf a, v8bf b) {
  return __builtin_shufflevector(a, b, 0,1,2,3,4,5,6,7,8,9,10,11,12,13,14,15);
}

// --- fused MGGP-RBF: S = X*Z^T via split-bf16 WMMA, epilogue does the kernel math ---
__global__ __launch_bounds__(256) void mggp_rbf_wmma_kernel(
    const float* __restrict__ X, const float* __restrict__ Z,
    const int* __restrict__ gX, const int* __restrict__ gZ,
    const float* __restrict__ E,
    const float* __restrict__ sigP, const float* __restrict__ lsP,
    const float* __restrict__ gdpP,
    const float* __restrict__ aa, const float* __restrict__ bb,
    float* __restrict__ out, int N, int M, int D, int G)
{
  __shared__ __bf16 sAhi[BM * LDT];
  __shared__ __bf16 sAlo[BM * LDT];
  __shared__ __bf16 sBhi[BN * LDT];
  __shared__ __bf16 sBlo[BN * LDT];
  __shared__ float  sAA[BM];
  __shared__ float  sBB[BN];
  __shared__ int    sGX[BM];
  __shared__ int    sGZ[BN];
  __shared__ float  sGt[16 * 16];

  const int tid  = threadIdx.x;
  const int lane = tid & 31;
  const int wave = tid >> 5;
  const int blockRow = blockIdx.y * BM;
  const int blockCol = blockIdx.x * BN;

  // group squared-distance lookup table (G<=16), replaces the [N,M] embedding GEMM
  if (tid < G * G) {
    const int a = tid / G, b = tid % G;
    float s = 0.f;
    for (int g = 0; g < G; ++g) { const float d = E[a*G+g] - E[b*G+g]; s = fmaf(d, d, s); }
    sGt[a * 16 + b] = s;
  }
  if (tid < BM) {
    const int r = (blockRow + tid < N) ? blockRow + tid : N - 1;
    sAA[tid] = aa[r]; sGX[tid] = gX[r];
  } else {
    const int t = tid - BM;
    const int c = (blockCol + t < M) ? blockCol + t : M - 1;
    sBB[t] = bb[c]; sGZ[t] = gZ[c];
  }

  // 4x2 wave grid; each wave computes a 32(M) x 64(N) patch = 2x4 16x16 tiles
  const int waveM = (wave >> 1) * 32;
  const int waveN = (wave & 1) * 64;
  const int half  = lane >> 4;   // selects K-phase per ISA wave32 WMMA layouts
  const int l15   = lane & 15;

  v8f acc[2][4] = {};

  const int lrow = (tid * 4) >> 5;  // staging: row within a 32-row slab
  const int lk   = (tid * 4) & 31;  // 4-aligned k offset
  const int kTiles = D / BK;        // D assumed multiple of 32 (D=512)

  for (int kt = 0; kt < kTiles; ++kt) {
    __syncthreads();  // WAR: previous iteration's fragment reads done
    const int k0 = kt * BK;
#pragma unroll
    for (int i = 0; i < 4; ++i) {
      const int r  = lrow + i * 32;
      const int rg = (blockRow + r < N) ? blockRow + r : N - 1;
      const int cg = (blockCol + r < M) ? blockCol + r : M - 1;
      const float4 vx = *(const float4*)(X + (size_t)rg * D + k0 + lk);
      const float4 vz = *(const float4*)(Z + (size_t)cg * D + k0 + lk);
      v4bf xh, xl, zh, zl;
      xh.x = (__bf16)vx.x; xl.x = (__bf16)(vx.x - (float)xh.x);
      xh.y = (__bf16)vx.y; xl.y = (__bf16)(vx.y - (float)xh.y);
      xh.z = (__bf16)vx.z; xl.z = (__bf16)(vx.z - (float)xh.z);
      xh.w = (__bf16)vx.w; xl.w = (__bf16)(vx.w - (float)xh.w);
      zh.x = (__bf16)vz.x; zl.x = (__bf16)(vz.x - (float)zh.x);
      zh.y = (__bf16)vz.y; zl.y = (__bf16)(vz.y - (float)zh.y);
      zh.z = (__bf16)vz.z; zl.z = (__bf16)(vz.z - (float)zh.z);
      zh.w = (__bf16)vz.w; zl.w = (__bf16)(vz.w - (float)zh.w);
      *(v4bf*)&sAhi[r * LDT + lk] = xh;
      *(v4bf*)&sAlo[r * LDT + lk] = xl;
      *(v4bf*)&sBhi[r * LDT + lk] = zh;
      *(v4bf*)&sBlo[r * LDT + lk] = zl;
    }
    __syncthreads();

    // A fragments (16x32 bf16): lanes 0-15 -> K {0-7,16-23}; lanes 16-31 -> K {8-15,24-31}
    v16bf Af[2][2];
#pragma unroll
    for (int mt = 0; mt < 2; ++mt) {
      const int base = (waveM + mt * 16 + l15) * LDT + half * 8;
      Af[mt][0] = cat8(*(const v8bf*)&sAhi[base], *(const v8bf*)&sAhi[base + 16]);
      Af[mt][1] = cat8(*(const v8bf*)&sAlo[base], *(const v8bf*)&sAlo[base + 16]);
    }
    // B fragments (32x16 bf16): lanes 0-15 -> K {0-7,16-23}; lanes 16-31 -> K {8-15,24-31} (col n = l15)
    v16bf Bf[4][2];
#pragma unroll
    for (int nt = 0; nt < 4; ++nt) {
      const int base = (waveN + nt * 16 + l15) * LDT + half * 8;
      Bf[nt][0] = cat8(*(const v8bf*)&sBhi[base], *(const v8bf*)&sBhi[base + 16]);
      Bf[nt][1] = cat8(*(const v8bf*)&sBlo[base], *(const v8bf*)&sBlo[base + 16]);
    }

    // split-bf16 f32 emulation: S += Ahi*Bhi + Ahi*Blo + Alo*Bhi
#pragma unroll
    for (int mt = 0; mt < 2; ++mt)
#pragma unroll
      for (int nt = 0; nt < 4; ++nt) {
        acc[mt][nt] = __builtin_amdgcn_wmma_f32_16x16x32_bf16(
            false, Af[mt][0], false, Bf[nt][0], (short)0, acc[mt][nt], false, false);
        acc[mt][nt] = __builtin_amdgcn_wmma_f32_16x16x32_bf16(
            false, Af[mt][0], false, Bf[nt][1], (short)0, acc[mt][nt], false, false);
        acc[mt][nt] = __builtin_amdgcn_wmma_f32_16x16x32_bf16(
            false, Af[mt][1], false, Bf[nt][0], (short)0, acc[mt][nt], false, false);
      }
  }

  // fused epilogue
  const float sigma   = *sigP;
  const float ls      = *lsP;
  const float agdp    = fabsf(*gdpP);
  const float inv_ls2 = 1.f / (ls * ls);
  const float sigma2  = sigma * sigma;
  const float halfp   = 0.5f * (float)D;

#pragma unroll
  for (int mt = 0; mt < 2; ++mt)
#pragma unroll
    for (int nt = 0; nt < 4; ++nt) {
      const int nl = waveN + nt * 16 + l15;
      const int gn = blockCol + nl;
      const float bbv = sBB[nl];
      const int   gzv = sGZ[nl];
#pragma unroll
      for (int j = 0; j < 8; ++j) {
        const int ml = waveM + mt * 16 + half * 8 + j;  // C layout: VGPR j, lane-half picks M or M+8
        const int gm = blockRow + ml;
        const float S    = acc[mt][nt][j];
        const float dist = sAA[ml] + bbv - 2.f * S;
        const float val  = 1.f / fmaf(agdp, sGt[sGX[ml] * 16 + gzv], 1.f);
        // sigma^2 * exp(-0.5*dist/ls^2*val) * val^(p/2) = sigma^2 * exp(arg)
        const float arg  = fmaf(-0.5f * inv_ls2 * val, dist, halfp * __logf(val));
        if (gm < N && gn < M) out[(size_t)gm * M + gn] = sigma2 * __expf(arg);
      }
    }
}

extern "C" void kernel_launch(void* const* d_in, const int* in_sizes, int n_in,
                              void* d_out, int out_size, void* d_ws, size_t ws_size,
                              hipStream_t stream) {
  const float* X   = (const float*)d_in[0];
  const float* Z   = (const float*)d_in[1];
  const int*   gX  = (const int*)d_in[2];
  const int*   gZ  = (const int*)d_in[3];
  const float* E   = (const float*)d_in[4];
  const float* sig = (const float*)d_in[5];
  const float* ls  = (const float*)d_in[6];
  const float* gdp = (const float*)d_in[7];

  const int N = in_sizes[2];
  const int M = in_sizes[3];
  const int D = in_sizes[0] / N;
  int G = 1; while (G * G < in_sizes[4]) ++G;

  float* aa = (float*)d_ws;      // ||x_i||^2  [N]
  float* bb = aa + N;            // ||z_j||^2  [M]

  row_sumsq_kernel<<<(N + 7) / 8, 256, 0, stream>>>(X, aa, N, D);
  row_sumsq_kernel<<<(M + 7) / 8, 256, 0, stream>>>(Z, bb, M, D);

  dim3 grid((M + BN - 1) / BN, (N + BM - 1) / BM);
  mggp_rbf_wmma_kernel<<<grid, 256, 0, stream>>>(X, Z, gX, gZ, E, sig, ls, gdp,
                                                 aa, bb, (float*)d_out, N, M, D, G);
}
